// SingleAttentionLongRangeProcessor_50173807952777
// MI455X (gfx1250) — compile-verified, matrix operand order checked by reading, not run on hardware
//
#include <hip/hip_runtime.h>
#include <hip/hip_bf16.h>

// ---------------------------------------------------------------------------
// Sliding-window attention for MI455X (gfx1250, wave32, WMMA).
//   B=2, S=2048, E=256, H=8, hd=32, window W=129 (off=64).
// Memory-bound problem (~25MB HBM traffic, ~2.7 GFLOP): all GEMM-like math is
// routed through v_wmma_f32_16x16x32_f16 (f16 inputs, f32 accumulate);
// intermediates kept in f16 to halve traffic. Everything fits in L2 (192MB).
// ---------------------------------------------------------------------------

typedef __attribute__((ext_vector_type(16))) _Float16     v16h;
typedef __attribute__((ext_vector_type(8)))  float        v8f;
typedef __attribute__((ext_vector_type(4)))  unsigned int v4u;

union F16x16 { v16h h; v4u u[2]; };

#define Bc 2
#define Sc 2048
#define Ec 256
#define Hc 8
#define HDc 32
#define Wc 129

// Loads a 16-half WMMA fragment: halves p[0..7] -> frag[0..7] (K = half16*8+i),
// halves p[16..23] -> frag[8..15] (K = 16 + half16*8 + i). p must be 16B aligned.
__device__ __forceinline__ v16h frag_ld(const _Float16* p) {
  F16x16 f;
  f.u[0] = *(const v4u*)(p);
  f.u[1] = *(const v4u*)(p + 16);
  return f.h;
}

__device__ __forceinline__ v8f wmma_f16(v16h a, v16h b, v8f c) {
  return __builtin_amdgcn_wmma_f32_16x16x32_f16(false, a, false, b, (short)0, c,
                                                false, false);
}

// --- Kernel 1: xh = f16(x + sinusoidal positional encoding) -----------------
__global__ __launch_bounds__(256) void prep_x_kernel(const float* __restrict__ x,
                                                     _Float16* __restrict__ xh) {
  int idx = blockIdx.x * 256 + threadIdx.x;        // < B*S*E
  int e = idx & (Ec - 1);
  int s = (idx >> 8) & (Sc - 1);
  int i2 = e >> 1;
  float div = __expf(-9.210340371976184f * (float)(2 * i2) / (float)Ec);
  float ang = (float)s * div;
  float pe = (e & 1) ? __cosf(ang) : __sinf(ang);
  xh[idx] = (_Float16)(x[idx] + pe);
}

// --- Kernel 2: Wt[w][n][k] = f16(W[k][n]) for Wq,Wk,Wv,Wo -------------------
__global__ __launch_bounds__(256) void prep_w_kernel(const float* __restrict__ Wq,
                                                     const float* __restrict__ Wk,
                                                     const float* __restrict__ Wv,
                                                     const float* __restrict__ Wo,
                                                     _Float16* __restrict__ wt) {
  int idx = blockIdx.x * 256 + threadIdx.x;        // < 4*E*E
  int w = idx >> 16;
  int rem = idx & 65535;
  int n = rem >> 8, k = rem & 255;
  const float* W = (w == 0) ? Wq : (w == 1) ? Wk : (w == 2) ? Wv : Wo;
  wt[idx] = (_Float16)W[k * Ec + n];               // wt layout: [w][n*256 + k]
}

// --- Kernel 3: Q/K/V projections. One 16x16 tile per wave, K=256 in 8 WMMAs.
// Q,K stored [B,H,S,32] f16; V stored transposed [B,H,32,S] f16 so the PV
// B-fragments in the attention kernel are contiguous 16B loads.
__global__ __launch_bounds__(256) void gemm_qkv_kernel(
    const _Float16* __restrict__ xh, const _Float16* __restrict__ wt,
    const float* __restrict__ bq, const float* __restrict__ bk,
    const float* __restrict__ bv,
    _Float16* __restrict__ Qb, _Float16* __restrict__ Kb,
    _Float16* __restrict__ Vt) {
  int z = blockIdx.z;                              // 0=Q 1=K 2=V
  const _Float16* w = wt + (size_t)z * Ec * Ec;
  const float* bias = (z == 0) ? bq : (z == 1) ? bk : bv;

  int lane = threadIdx.x & 31;
  int wave = threadIdx.x >> 5;
  int tile = blockIdx.x * 8 + wave;                // 0..4095
  int tm = tile >> 4;                              // 256 row tiles
  int tn = tile & 15;                              // 16  col tiles
  int r16 = lane & 15;
  int half16 = lane >> 4;

  int arow = tm * 16 + r16;
  int bcol = tn * 16 + r16;
  const _Float16* ap = xh + (size_t)arow * Ec + half16 * 8;
  const _Float16* bp = w + (size_t)bcol * Ec + half16 * 8;

  v8f acc = {};
#pragma unroll
  for (int k0 = 0; k0 < Ec; k0 += 32) {
    __builtin_prefetch(ap + k0 + 64, 0, 1);        // global_prefetch_b8
    v16h a = frag_ld(ap + k0);
    v16h b = frag_ld(bp + k0);
    acc = wmma_f16(a, b, acc);
  }

  int n = tn * 16 + r16;                           // D element: N = lane&15
  int h = n >> 5, dd = n & 31;
  float bn = bias[n];
#pragma unroll
  for (int g = 0; g < 8; ++g) {                    // M = g + 8*half16
    int m = tm * 16 + g + 8 * half16;
    int b_ = m >> 11, s = m & (Sc - 1);
    _Float16 val = (_Float16)(acc[g] + bn);
    if (z == 2)
      Vt[(((size_t)(b_ * Hc + h) * HDc) + dd) * Sc + s] = val;
    else {
      _Float16* o = (z == 0) ? Qb : Kb;
      o[(((size_t)(b_ * Hc + h) * Sc) + s) * HDc + dd] = val;
    }
  }
}

// --- Kernel 4: windowed attention. One 16-row s-tile per wave. --------------
// scores: 9 WMMAs (Q 16x32 x K^T 32x16 per j-tile; hd=32 = exactly one K-step)
// softmax in f32 via LDS (writes attn f32 to d_out, packs f16 probs in place)
// ctx = P(16x160) x Vwin(160x32): 2 n-tiles x 5 k-chunks = 10 WMMAs.
__global__ __launch_bounds__(128) void attn_kernel(
    const _Float16* __restrict__ Qb, const _Float16* __restrict__ Kb,
    const _Float16* __restrict__ Vt, float* __restrict__ attn_out,
    _Float16* __restrict__ ctx) {
  __shared__ float smem[4][16][144];               // 36.9 KB / workgroup

  int lane = threadIdx.x & 31;
  int wave = threadIdx.x >> 5;
  int gid = blockIdx.x * 4 + wave;                 // 0..2047
  int bh = gid >> 7;                               // b*H + h (16 values)
  int s0 = (gid & 127) * 16;
  int r16 = lane & 15;
  int half16 = lane >> 4;
  int jlo = s0 - 64;

  float (*sw)[144] = smem[wave];

  // ---- scores strip: j in [jlo, jlo+143], 9 tiles of 16 columns
  v16h aq = frag_ld(Qb + ((size_t)bh * Sc + (s0 + r16)) * HDc + half16 * 8);
#pragma unroll
  for (int c = 0; c < 9; ++c) {
    int j = jlo + c * 16 + r16;
    int jc = j < 0 ? 0 : (j > Sc - 1 ? Sc - 1 : j);   // masked later
    v16h bk_ = frag_ld(Kb + ((size_t)bh * Sc + jc) * HDc + half16 * 8);
    v8f d = {};
    d = wmma_f16(aq, bk_, d);
#pragma unroll
    for (int g = 0; g < 8; ++g)
      sw[g + 8 * half16][c * 16 + r16] = d[g];
  }
  __syncthreads();

  // ---- masked softmax, one row per lane (lanes 0..15)
  if (lane < 16) {
    int r = lane;
    int s = s0 + r;
    float* srow = sw[r];
    const float scale = 0.17677669529663687f;      // 1/sqrt(32)
    float mx = -1e30f;
    for (int c = 0; c < 144; ++c) {
      int j = jlo + c, w = c - r;
      if (w >= 0 && w <= 128 && j >= 0 && j < Sc) {
        float sc = srow[c] * scale;
        mx = sc > mx ? sc : mx;
      }
    }
    float sum = 0.f;
    for (int c = 0; c < 144; ++c) {
      int j = jlo + c, w = c - r;
      float e = (w >= 0 && w <= 128 && j >= 0 && j < Sc)
                    ? __expf(srow[c] * scale - mx) : 0.f;
      srow[c] = e;
      sum += e;
    }
    float inv = 1.f / sum;
    _Float16* hrow = (_Float16*)srow;              // pack f16 probs in place
    float* arow = attn_out + ((size_t)bh * Sc + s) * Wc;
    for (int c = 0; c < 144; ++c) {                // ascending: 2c <= 4c safe
      float p = srow[c] * inv;
      int w = c - r;
      if (w >= 0 && w <= 128) arow[w] = p;         // attn output (f32)
      hrow[c] = (_Float16)p;
    }
    for (int c = 144; c < 160; ++c) hrow[c] = (_Float16)0.f;  // zero K-pad
  }
  __syncthreads();

  // ---- ctx = P @ Vwin  (probs rows are 288 halves apart in LDS)
  const _Float16* hsm = (const _Float16*)&smem[wave][0][0];
  int b_ = bh >> 3, h = bh & 7;
#pragma unroll
  for (int nt = 0; nt < 2; ++nt) {
    int dd = nt * 16 + r16;
    const _Float16* vrow = Vt + ((size_t)bh * HDc + dd) * Sc;
    v8f acc = {};
#pragma unroll
    for (int c5 = 0; c5 < 5; ++c5) {
      v16h ap = frag_ld(hsm + r16 * 288 + c5 * 32 + half16 * 8);
      int jb = jlo + c5 * 32 + half16 * 8;
      // groups of 8 are entirely in/out of range (all indices mult. of 8),
      // so clamping the base is safe: out-of-range groups multiply P==0.
      int jb0 = jb < 0 ? 0 : (jb > Sc - 8 ? Sc - 8 : jb);
      int jb1 = (jb + 16) < 0 ? 0 : ((jb + 16) > Sc - 8 ? Sc - 8 : jb + 16);
      F16x16 fb;
      fb.u[0] = *(const v4u*)(vrow + jb0);
      fb.u[1] = *(const v4u*)(vrow + jb1);
      acc = wmma_f16(ap, fb.h, acc);
    }
#pragma unroll
    for (int g = 0; g < 8; ++g) {
      int s = s0 + g + 8 * half16;
      ctx[((size_t)b_ * Sc + s) * Ec + h * HDc + dd] = (_Float16)acc[g];
    }
  }
}

// --- Kernel 5: out = ctx(f16) @ Wo + bo, f32 output -------------------------
__global__ __launch_bounds__(256) void gemm_out_kernel(
    const _Float16* __restrict__ ctx, const _Float16* __restrict__ wto,
    const float* __restrict__ bo, float* __restrict__ out) {
  int lane = threadIdx.x & 31;
  int wave = threadIdx.x >> 5;
  int tile = blockIdx.x * 8 + wave;
  int tm = tile >> 4, tn = tile & 15;
  int r16 = lane & 15, half16 = lane >> 4;

  const _Float16* ap = ctx + (size_t)(tm * 16 + r16) * Ec + half16 * 8;
  const _Float16* bp = wto + (size_t)(tn * 16 + r16) * Ec + half16 * 8;

  v8f acc = {};
#pragma unroll
  for (int k0 = 0; k0 < Ec; k0 += 32) {
    __builtin_prefetch(ap + k0 + 64, 0, 1);
    v16h a = frag_ld(ap + k0);
    v16h b = frag_ld(bp + k0);
    acc = wmma_f16(a, b, acc);
  }
  int n = tn * 16 + r16;
  float bn = bo[n];
#pragma unroll
  for (int g = 0; g < 8; ++g) {
    int m = tm * 16 + g + 8 * half16;
    out[(size_t)m * Ec + n] = acc[g] + bn;
  }
}

extern "C" void kernel_launch(void* const* d_in, const int* in_sizes, int n_in,
                              void* d_out, int out_size, void* d_ws,
                              size_t ws_size, hipStream_t stream) {
  const float* x  = (const float*)d_in[0];
  const float* Wq = (const float*)d_in[1];
  const float* bq = (const float*)d_in[2];
  const float* Wk = (const float*)d_in[3];
  const float* bk = (const float*)d_in[4];
  const float* Wv = (const float*)d_in[5];
  const float* bv = (const float*)d_in[6];
  const float* Wo = (const float*)d_in[7];
  const float* bo = (const float*)d_in[8];
  float* out = (float*)d_out;                       // [B,S,E] then attn [B,H,S,W]
  float* attn = out + (size_t)Bc * Sc * Ec;

  // workspace layout (8.5 MB total, 16B aligned):
  char* ws = (char*)d_ws;
  _Float16* xh = (_Float16*)ws;                     // 2 MB (reused as ctx later)
  _Float16* wt = (_Float16*)(ws + (2u << 20));      // 0.5 MB (4 x 256x256 f16)
  _Float16* Qb = (_Float16*)(ws + (2u << 20) + (512u << 10));  // 2 MB
  _Float16* Kb = Qb + (size_t)Bc * Hc * Sc * HDc;              // 2 MB
  _Float16* Vt = Kb + (size_t)Bc * Hc * Sc * HDc;              // 2 MB
  _Float16* ctx = xh;   // xh dead after gemm_qkv; reuse for ctx

  prep_x_kernel<<<(Bc * Sc * Ec) / 256, 256, 0, stream>>>(x, xh);
  prep_w_kernel<<<(4 * Ec * Ec) / 256, 256, 0, stream>>>(Wq, Wk, Wv, Wo, wt);
  gemm_qkv_kernel<<<dim3(512, 1, 3), 256, 0, stream>>>(xh, wt, bq, bk, bv,
                                                       Qb, Kb, Vt);
  attn_kernel<<<512, 128, 0, stream>>>(Qb, Kb, Vt, attn, ctx);
  gemm_out_kernel<<<512, 256, 0, stream>>>(ctx, wt + 3 * Ec * Ec, bo, out);
}
